// NewAttention_82377472737869
// MI455X (gfx1250) — compile-verified, matrix operand order checked by reading, not run on hardware
//
#include <hip/hip_runtime.h>
#include <hip/hip_bf16.h>

// ---------------------------------------------------------------------------
// MI455X (gfx1250) fused attention:
//   X @ Wqkv + b  -> RoPE(q,k) -> flash softmax(QK^T/8 + bias) V -> ctx @ Wo + bo
// All matmuls via v_wmma_f32_16x16x32_bf16 (wave32). Shared tiles (GEMM B,
// attention K/V) staged through double-buffered LDS with CDNA5 async copies
// (global_load_async_to_lds_b128 + s_wait_asynccnt) to kill the 8x duplicate
// global traffic across the 8 waves of each workgroup.
// ---------------------------------------------------------------------------

typedef __bf16 v16bf __attribute__((ext_vector_type(16)));
typedef __bf16 v8bf  __attribute__((ext_vector_type(8)));
typedef float  v8f   __attribute__((ext_vector_type(8)));
typedef int    vsi4  __attribute__((vector_size(16)));   // int4 for async builtin

#define Bc 2
#define Sc 2048
#define Dc 1024
#define Hc 16
#define HDc 64

#if defined(__has_builtin)
#if __has_builtin(__builtin_amdgcn_global_load_async_to_lds_b128)
#define USE_ASYNC_LDS 1
#endif
#endif
#ifndef USE_ASYNC_LDS
#define USE_ASYNC_LDS 0
#endif

#define AS1 __attribute__((address_space(1)))
#define AS3 __attribute__((address_space(3)))

// Copy 16 bytes global -> LDS. Async DMA path on CDNA5 if available.
__device__ __forceinline__ void stage16(const __bf16* g, __bf16* l) {
#if USE_ASYNC_LDS
  // Generic LDS addresses carry the LDS offset in the low 32 bits (ISA 10.2),
  // AS3 pointers are 32-bit, so the integer round-trip is exact.
  __builtin_amdgcn_global_load_async_to_lds_b128(
      (AS1 vsi4*)(uintptr_t)g, (AS3 vsi4*)(unsigned int)(uintptr_t)l, 0, 0);
#else
  *(v8bf*)l = *(const v8bf*)g;
#endif
}

// Wait for outstanding async stages, then publish to the workgroup.
__device__ __forceinline__ void wait_stage() {
#if USE_ASYNC_LDS
#if __has_builtin(__builtin_amdgcn_s_wait_asynccnt)
  __builtin_amdgcn_s_wait_asynccnt(0);
#else
  asm volatile("s_wait_asynccnt 0" ::: "memory");
#endif
#endif
  __syncthreads();
}

__device__ __forceinline__ v16bf load2(const __bf16* p0, const __bf16* p1) {
  union { v16bf v; v8bf h[2]; } u;
  u.h[0] = *(const v8bf*)p0;
  u.h[1] = *(const v8bf*)p1;
  return u.v;
}

__device__ __forceinline__ v8f wmma_bf16(v16bf a, v16bf b, v8f c) {
  return __builtin_amdgcn_wmma_f32_16x16x32_bf16(false, a, false, b, (short)0, c,
                                                 false, false);
}

// ---------------------------------------------------------------------------
// Prepass: f32 -> bf16 convert, and transpose+convert for weights
// ---------------------------------------------------------------------------
__global__ void __launch_bounds__(256) cvt_bf16_kernel(const float* __restrict__ src,
                                                       __bf16* __restrict__ dst, int n) {
  int i = blockIdx.x * blockDim.x + threadIdx.x;
  if (i < n) dst[i] = (__bf16)src[i];
}

__global__ void __launch_bounds__(256) transpose_bf16_kernel(const float* __restrict__ src,
                                                             __bf16* __restrict__ dst,
                                                             int R, int C) {
  int i = blockIdx.x * blockDim.x + threadIdx.x;
  if (i < R * C) {
    int r = i / C, c = i - r * C;
    dst[(size_t)c * R + r] = (__bf16)src[i];  // dst is (C,R) = src^T
  }
}

// ---------------------------------------------------------------------------
// Kernel 1: qkv = Xb @ WqkvT^T + bqkv; RoPE on q,k; scatter Q,K (B,H,S,HD),
// V transposed (B,H,HD,S).  Wave tile: 16 rows x 64 cols (one head).
// B tile (64x32, shared by all 8 waves) async-staged through LDS, 2 buffers.
// Grid (4096/128, 3072/64), block 256 (8 waves).
// ---------------------------------------------------------------------------
__global__ void __launch_bounds__(256) qkv_rope_kernel(
    const __bf16* __restrict__ Xb,     // (B*S, D)
    const __bf16* __restrict__ WqkvT,  // (3D, D)
    const float*  __restrict__ bqkv,   // (3D)
    const float*  __restrict__ cosp,   // (S, HD)
    const float*  __restrict__ sinp,   // (S, HD)
    __bf16* __restrict__ Qb,           // (B,H,S,HD)
    __bf16* __restrict__ Kb,           // (B,H,S,HD)
    __bf16* __restrict__ VT)           // (B,H,HD,S)
{
  __shared__ __align__(16) __bf16 bst[2][64][32];   // 8 KB

  const int tid   = threadIdx.x;
  const int lane  = tid & 31;
  const int wave  = tid >> 5;
  const int ln    = lane & 15;
  const int koff  = (lane >> 4) * 8;
  const int m0    = blockIdx.x * 128 + wave * 16;
  const int n0    = blockIdx.y * 64;

  const int srow = tid >> 2, schunk = (tid & 3) * 8;  // 64 rows x 4 chunks
  const __bf16* bsrc = WqkvT + (size_t)(n0 + srow) * Dc + schunk;

  stage16(bsrc, &bst[0][srow][schunk]);  // tile k=0

  v8f acc[4] = {};
  const __bf16* arow = Xb + (size_t)(m0 + ln) * Dc;
  for (int k = 0; k < Dc; k += 32) {
    const int buf = (k >> 5) & 1;
    wait_stage();
    if (k + 32 < Dc) stage16(bsrc + k + 32, &bst[buf ^ 1][srow][schunk]);
    v16bf a = load2(arow + k + koff, arow + k + 16 + koff);
#pragma unroll
    for (int j = 0; j < 4; ++j) {
      v16bf b = load2(&bst[buf][j * 16 + ln][koff], &bst[buf][j * 16 + ln][16 + koff]);
      acc[j] = wmma_bf16(a, b, acc[j]);
    }
  }

  const int region = n0 >> 10;            // 0=q 1=k 2=v
  const int h      = (n0 & 1023) >> 6;
  const int b      = m0 >> 11;
  const int bh     = b * Hc + h;
  const int sbase  = m0 & (Sc - 1);

  float vals[4][8];
#pragma unroll
  for (int j = 0; j < 4; ++j) {
    float bv = bqkv[n0 + j * 16 + ln];
#pragma unroll
    for (int e = 0; e < 8; ++e) vals[j][e] = acc[j][e] + bv;
  }

  if (region == 2) {
#pragma unroll
    for (int j = 0; j < 4; ++j) {
      int d = j * 16 + ln;
      __bf16* vp = VT + ((size_t)bh * HDc + d) * Sc + sbase + koff;
#pragma unroll
      for (int e = 0; e < 8; ++e) vp[e] = (__bf16)vals[j][e];
    }
  } else {
    __bf16* dstbuf = (region == 0) ? Qb : Kb;
#pragma unroll
    for (int e = 0; e < 8; ++e) {
      int s = sbase + e + koff;  // C layout: rows m = e (+8 for hi lanes)
#pragma unroll
      for (int j = 0; j < 4; ++j) {
        int d = j * 16 + ln;
        float cv = cosp[s * HDc + d];
        float sv = sinp[s * HDc + d];
        float rot = (j < 2) ? -vals[j + 2][e] : vals[j - 2][e];
        dstbuf[((size_t)bh * Sc + s) * HDc + d] = (__bf16)(vals[j][e] * cv + rot * sv);
      }
    }
  }
}

// ---------------------------------------------------------------------------
// Kernel 2: flash attention. One wave per 16-query block, 32 keys per step.
// Scores computed transposed (S^T = K * Q^T) so softmax reduces along VGPRs
// (+ one shfl_xor(16) for the lane-pair halves). K and V^T tiles are shared
// by all 8 waves -> async-staged through double-buffered LDS. P is staged
// per-wave through LDS to reshape into the WMMA B layout; O^T = V^T * P^T.
// Grid (B*H, S/128), block 256 (8 waves).
// ---------------------------------------------------------------------------
__global__ void __launch_bounds__(256) attn_kernel(
    const __bf16* __restrict__ Qb,     // (B,H,S,HD)
    const __bf16* __restrict__ Kb,     // (B,H,S,HD)
    const __bf16* __restrict__ VT,     // (B,H,HD,S)
    const float*  __restrict__ bias,   // (B,1,S,S)
    __bf16* __restrict__ ctx)          // (B,S,H,HD)
{
  __shared__ __align__(16) __bf16 kst[2][32][64];   // 8 KB, K rows tile
  __shared__ __align__(16) __bf16 vst[2][64][32];   // 8 KB, V^T rows tile
  __shared__ __align__(16) __bf16 plds[8][16][32];  // 8 KB, per-wave P tile

  const int tid  = threadIdx.x;
  const int lane = tid & 31;
  const int wave = tid >> 5;
  const int ln   = lane & 15;
  const int koff = (lane >> 4) * 8;
  const int bh   = blockIdx.x;
  const int b    = bh >> 4;
  const int h    = bh & 15;
  const int q0   = blockIdx.y * 128 + wave * 16;
  const int q    = q0 + ln;

  // staging maps: K tile 32 rows x 8 chunks, V tile 64 rows x 4 chunks
  const int krow = tid >> 3, kch = (tid & 7) * 8;
  const int vrow = tid >> 2, vch = (tid & 3) * 8;
  const __bf16* ksrc = Kb + ((size_t)bh * Sc + krow) * HDc + kch;
  const __bf16* vsrc = VT + ((size_t)bh * HDc + vrow) * Sc + vch;

  stage16(ksrc, &kst[0][krow][kch]);
  stage16(vsrc, &vst[0][vrow][vch]);

  // Q as B-operand (column q of Q^T == row q of Q): two 32-deep chunks of HD
  const __bf16* qrow = Qb + ((size_t)bh * Sc + q) * HDc;
  v16bf qb1 = load2(qrow + koff,      qrow + 16 + koff);
  v16bf qb2 = load2(qrow + 32 + koff, qrow + 48 + koff);

  v8f o[4] = {};
  float mrun = -1e30f, lrun = 0.f;
  const float* biasb = bias + (size_t)b * Sc * Sc;

  for (int k0 = 0; k0 < Sc; k0 += 32) {
    const int buf = (k0 >> 5) & 1;
    wait_stage();
    if (k0 + 32 < Sc) {
      stage16(ksrc + (size_t)(k0 + 32) * HDc, &kst[buf ^ 1][krow][kch]);
      stage16(vsrc + k0 + 32, &vst[buf ^ 1][vrow][vch]);
    }

    float sc[2][8];
    float pmax = -1e30f;
#pragma unroll
    for (int ks = 0; ks < 2; ++ks) {
      const __bf16* kr = &kst[buf][ks * 16 + ln][0];
      v16bf ka1 = load2(kr + koff,      kr + 16 + koff);
      v16bf ka2 = load2(kr + 32 + koff, kr + 48 + koff);
      v8f c = {};
      c = wmma_bf16(ka1, qb1, c);
      c = wmma_bf16(ka2, qb2, c);
#pragma unroll
      for (int e = 0; e < 8; ++e) {
        int key = k0 + ks * 16 + e + koff;  // C rows = keys
        float v = c[e] * 0.125f + biasb[(size_t)q * Sc + key];
        sc[ks][e] = v;
        pmax = fmaxf(pmax, v);
      }
    }
    pmax = fmaxf(pmax, __shfl_xor(pmax, 16, 32));
    float mnew  = fmaxf(mrun, pmax);
    float alpha = __expf(mrun - mnew);
    float psum  = 0.f;
#pragma unroll
    for (int ks = 0; ks < 2; ++ks)
#pragma unroll
      for (int e = 0; e < 8; ++e) {
        float p = __expf(sc[ks][e] - mnew);
        psum += p;
        plds[wave][ln][ks * 16 + e + koff] = (__bf16)p;  // P[q][k]
      }
    psum += __shfl_xor(psum, 16, 32);
    lrun = lrun * alpha + psum;
    mrun = mnew;
#pragma unroll
    for (int j = 0; j < 4; ++j) o[j] = o[j] * alpha;

    asm volatile("s_wait_dscnt 0" ::: "memory");  // P stores visible to ds reads
    v16bf pB = load2(&plds[wave][ln][koff], &plds[wave][ln][16 + koff]);
#pragma unroll
    for (int dsb = 0; dsb < 4; ++dsb) {
      const __bf16* vr = &vst[buf][dsb * 16 + ln][0];
      v16bf va = load2(vr + koff, vr + 16 + koff);
      o[dsb] = wmma_bf16(va, pB, o[dsb]);  // O^T += V^T * P^T
    }
  }

  float invl = (lrun > 0.f) ? (1.f / lrun) : 0.f;
#pragma unroll
  for (int dsb = 0; dsb < 4; ++dsb)
#pragma unroll
    for (int e = 0; e < 8; ++e) {
      int d = dsb * 16 + e + koff;  // O^T rows = head dim
      ctx[(((size_t)b * Sc + q) * Hc + h) * HDc + d] = (__bf16)(o[dsb][e] * invl);
    }
}

// ---------------------------------------------------------------------------
// Kernel 3: out = ctx @ Wo + bo (f32 output). Wave tile 16x64, B tile staged
// through double-buffered LDS. Grid (4096/128, 1024/64), block 256.
// ---------------------------------------------------------------------------
__global__ void __launch_bounds__(256) out_gemm_kernel(
    const __bf16* __restrict__ ctx,   // (B*S, D)
    const __bf16* __restrict__ WoT,   // (D, D) = Wo^T
    const float*  __restrict__ bo,    // (D)
    float* __restrict__ out)          // (B*S, D)
{
  __shared__ __align__(16) __bf16 bst[2][64][32];   // 8 KB

  const int tid  = threadIdx.x;
  const int lane = tid & 31;
  const int wave = tid >> 5;
  const int ln   = lane & 15;
  const int koff = (lane >> 4) * 8;
  const int m0   = blockIdx.x * 128 + wave * 16;
  const int n0   = blockIdx.y * 64;

  const int srow = tid >> 2, schunk = (tid & 3) * 8;
  const __bf16* bsrc = WoT + (size_t)(n0 + srow) * Dc + schunk;

  stage16(bsrc, &bst[0][srow][schunk]);

  v8f acc[4] = {};
  const __bf16* arow = ctx + (size_t)(m0 + ln) * Dc;
  for (int k = 0; k < Dc; k += 32) {
    const int buf = (k >> 5) & 1;
    wait_stage();
    if (k + 32 < Dc) stage16(bsrc + k + 32, &bst[buf ^ 1][srow][schunk]);
    v16bf a = load2(arow + k + koff, arow + k + 16 + koff);
#pragma unroll
    for (int j = 0; j < 4; ++j) {
      v16bf b = load2(&bst[buf][j * 16 + ln][koff], &bst[buf][j * 16 + ln][16 + koff]);
      acc[j] = wmma_bf16(a, b, acc[j]);
    }
  }
#pragma unroll
  for (int j = 0; j < 4; ++j) {
    float bv = bo[n0 + j * 16 + ln];
#pragma unroll
    for (int e = 0; e < 8; ++e)
      out[(size_t)(m0 + e + koff) * Dc + n0 + j * 16 + ln] = acc[j][e] + bv;
  }
}

// ---------------------------------------------------------------------------
extern "C" void kernel_launch(void* const* d_in, const int* in_sizes, int n_in,
                              void* d_out, int out_size, void* d_ws, size_t ws_size,
                              hipStream_t stream) {
  const float* hidden = (const float*)d_in[0];  // (B,S,D)
  const float* abias  = (const float*)d_in[1];  // (B,1,S,S)
  const float* cosp   = (const float*)d_in[2];  // (1,S,1,HD)
  const float* sinp   = (const float*)d_in[3];  // (1,S,1,HD)
  const float* Wqkv   = (const float*)d_in[4];  // (D,3D)
  const float* bqkv   = (const float*)d_in[5];  // (3D)
  const float* Wo     = (const float*)d_in[6];  // (D,D)
  const float* bo     = (const float*)d_in[7];  // (D)
  float* out = (float*)d_out;

  char* ws = (char*)d_ws;
  __bf16* Xb    = (__bf16*)(ws);                         // 8 MB  (B*S, D) bf16
  __bf16* WqkvT = (__bf16*)(ws + ((size_t)8  << 20));    // 6 MB  (3D, D)
  __bf16* WoT   = (__bf16*)(ws + ((size_t)14 << 20));    // 2 MB  (D, D)
  __bf16* Qb    = (__bf16*)(ws + ((size_t)16 << 20));    // 8 MB  (B,H,S,HD)
  __bf16* Kb    = (__bf16*)(ws + ((size_t)24 << 20));    // 8 MB  (B,H,S,HD)
  __bf16* VT    = (__bf16*)(ws + ((size_t)32 << 20));    // 8 MB  (B,H,HD,S)
  __bf16* ctxb  = Xb;  // alias: Xb dead after kernel 1   total 40 MB

  const int nX = Bc * Sc * Dc;  // 4194304
  cvt_bf16_kernel<<<(nX + 255) / 256, 256, 0, stream>>>(hidden, Xb, nX);
  transpose_bf16_kernel<<<(Dc * 3 * Dc + 255) / 256, 256, 0, stream>>>(Wqkv, WqkvT, Dc, 3 * Dc);
  transpose_bf16_kernel<<<(Dc * Dc + 255) / 256, 256, 0, stream>>>(Wo, WoT, Dc, Dc);

  qkv_rope_kernel<<<dim3((Bc * Sc) / 128, (3 * Dc) / 64), 256, 0, stream>>>(
      Xb, WqkvT, bqkv, cosp, sinp, Qb, Kb, VT);

  attn_kernel<<<dim3(Bc * Hc, Sc / 128), 256, 0, stream>>>(Qb, Kb, VT, abias, ctxb);

  out_gemm_kernel<<<dim3((Bc * Sc) / 128, Dc / 64), 256, 0, stream>>>(ctxb, WoT, bo, out);
}
